// GpsaRoIPool_77713138254111
// MI455X (gfx1250) — matrix operand order repaired and checked by reading, not compiled, not run on hardware
//
#include <hip/hip_runtime.h>
#include <hip/hip_bf16.h>
#include <math.h>

typedef __attribute__((ext_vector_type(16))) __bf16 v16bf;
typedef __attribute__((ext_vector_type(8)))  __bf16 v8bf;
typedef __attribute__((ext_vector_type(8)))  float  v8f;

#define CCH    256
#define FH     200
#define FW     200
#define HW     (FH * FW)       // 40000
#define NQ     49
#define NHEADS 4
#define HDIM   64

// ---------------- WMMA helpers (CDNA5 bf16 16x16x32) ----------------

static __device__ __forceinline__ v8f wmma_bf16(v16bf a, v16bf b, v8f c) {
  // (neg_a, A, neg_b, B, c_mod, C, reuse_a, reuse_b)
  return __builtin_amdgcn_wmma_f32_16x16x32_bf16(false, a, false, b, (short)0, c,
                                                 false, false);
}

// A-matrix per-lane pack from a row-major [M][K] bf16 row pointer.
// lanes 0-15 (hi=0): K = kb+0..15 ; lanes 16-31 (hi=1): K = kb+16..31
static __device__ __forceinline__ v16bf packA_bf(const __bf16* row, int kb, int hi) {
  return *(const v16bf*)(row + kb + 16 * hi);
}

// B-matrix per-lane pack: column n = lane%16, 16 contiguous K per lane half.
static __device__ __forceinline__ v16bf packB_bf(const __bf16* row, int kb, int hi) {
  return *(const v16bf*)(row + kb + 16 * hi);
}

// Same as packB_bf but converting from an f32 source row (weights).
static __device__ __forceinline__ v16bf packB_f32(const float* chunk) {
  const float4* p = (const float4*)chunk;
  float4 f0 = p[0], f1 = p[1], f2 = p[2], f3 = p[3];
  v16bf r;
  r[0]  = (__bf16)f0.x; r[1]  = (__bf16)f0.y; r[2]  = (__bf16)f0.z; r[3]  = (__bf16)f0.w;
  r[4]  = (__bf16)f1.x; r[5]  = (__bf16)f1.y; r[6]  = (__bf16)f1.z; r[7]  = (__bf16)f1.w;
  r[8]  = (__bf16)f2.x; r[9]  = (__bf16)f2.y; r[10] = (__bf16)f2.z; r[11] = (__bf16)f2.w;
  r[12] = (__bf16)f3.x; r[13] = (__bf16)f3.y; r[14] = (__bf16)f3.z; r[15] = (__bf16)f3.w;
  return r;
}

static __device__ __forceinline__ float wredMax(float v) {
#pragma unroll
  for (int off = 16; off > 0; off >>= 1) v = fmaxf(v, __shfl_xor(v, off, 32));
  return v;
}
static __device__ __forceinline__ float wredSum(float v) {
#pragma unroll
  for (int off = 16; off > 0; off >>= 1) v += __shfl_xor(v, off, 32);
  return v;
}

// ---------------- Kernel A: position softmax, balance, queries->bf16 ----------------

__global__ __launch_bounds__(256) void prep_kernel(
    const float* __restrict__ pw,       // [49,14,14]
    const float* __restrict__ queries,  // [49,256]
    const float* __restrict__ braw,     // [1]
    float* __restrict__ pos_attn,       // [49,256]
    __bf16* __restrict__ qb,            // [64,256] (rows 49..63 zero)
    float* __restrict__ balance_out) {
  __shared__ float sred[8];
  __shared__ float bcast;
  const int tid = threadIdx.x, lane = tid & 31, wid = tid >> 5;
  if (tid == 0) {
    float b = (braw[0] + 3.0f) * (1.0f / 6.0f);        // hardsigmoid
    *balance_out = fminf(fmaxf(b, 0.0f), 1.0f);
  }
  // F.interpolate(14->16, bilinear, align_corners=False) sample coords
  const int iy = tid >> 4, ix = tid & 15;
  float sy = fmaxf(((float)iy + 0.5f) * (14.0f / 16.0f) - 0.5f, 0.0f);
  float sx = fmaxf(((float)ix + 0.5f) * (14.0f / 16.0f) - 0.5f, 0.0f);
  int y0 = (int)floorf(sy); int y1 = min(y0 + 1, 13); float wy = sy - (float)y0;
  int x0 = (int)floorf(sx); int x1 = min(x0 + 1, 13); float wx = sx - (float)x0;

  for (int q = 0; q < NQ; ++q) {
    const float* img = pw + q * 196;
    float r0 = img[y0 * 14 + x0] * (1.0f - wy) + img[y1 * 14 + x0] * wy;
    float r1 = img[y0 * 14 + x1] * (1.0f - wy) + img[y1 * 14 + x1] * wy;
    float v = r0 * (1.0f - wx) + r1 * wx;

    float m = wredMax(v);
    if (lane == 0) sred[wid] = m;
    __syncthreads();
    if (tid == 0) {
      float mm = sred[0];
      for (int j = 1; j < 8; ++j) mm = fmaxf(mm, sred[j]);
      bcast = mm;
    }
    __syncthreads();
    float e = expf(v - bcast);
    float s = wredSum(e);
    __syncthreads();
    if (lane == 0) sred[wid] = s;
    __syncthreads();
    if (tid == 0) {
      float ss = 0.0f;
      for (int j = 0; j < 8; ++j) ss += sred[j];
      bcast = ss;
    }
    __syncthreads();
    pos_attn[q * 256 + tid] = e / bcast;
    __syncthreads();
  }
  for (int idx = tid; idx < 64 * 256; idx += 256) {
    int row = idx >> 8;
    float v = (row < NQ) ? queries[idx] : 0.0f;
    qb[idx] = (__bf16)v;
  }
}

// ---------------- Kernel B: fused 1x1 convs k,v -> NHWC bf16 via WMMA ----------------
// Grid: 2 batches * 625 tiles of 64 pixels. Block: 256 threads = 8 waves, each wave
// owns 32 output channels, computes both k and v for a 64x32 output tile.

__global__ __launch_bounds__(256) void conv_kv_kernel(
    const float* __restrict__ feat,                              // [2,256,200,200]
    const float* __restrict__ wk, const float* __restrict__ bk,  // [256,256],[256]
    const float* __restrict__ wv, const float* __restrict__ bv,
    __bf16* __restrict__ kout, __bf16* __restrict__ vout) {      // [2,40000,256] bf16
  __shared__ __bf16 ldsA[64 * 32];  // [m_local][k_local] bf16
  const int blk = blockIdx.x;
  const int b = blk / 625;
  const int m0 = (blk % 625) * 64;
  const int tid = threadIdx.x;
  const int lane = tid & 31, wid = tid >> 5;
  const int hi = lane >> 4, ln = lane & 15;
  const int n0 = wid * 32;

  v8f acc[2][2][4];  // [k|v][nt][mt]
  const v8f vzero = {0, 0, 0, 0, 0, 0, 0, 0};
#pragma unroll
  for (int a = 0; a < 2; ++a)
#pragma unroll
    for (int nt = 0; nt < 2; ++nt)
#pragma unroll
      for (int mt = 0; mt < 4; ++mt) acc[a][nt][mt] = vzero;

  const int cl = tid >> 3;       // local input channel 0..31
  const int ml = (tid & 7) * 8;  // local pixel 0,8,..56

  for (int kk = 0; kk < 8; ++kk) {
    const int kb = kk * 32;
    __syncthreads();
    {
      const float* src = feat + ((size_t)b * CCH + (size_t)(kb + cl)) * HW + m0 + ml;
      float4 f0 = *(const float4*)src;
      float4 f1 = *(const float4*)(src + 4);
      ldsA[(ml + 0) * 32 + cl] = (__bf16)f0.x;
      ldsA[(ml + 1) * 32 + cl] = (__bf16)f0.y;
      ldsA[(ml + 2) * 32 + cl] = (__bf16)f0.z;
      ldsA[(ml + 3) * 32 + cl] = (__bf16)f0.w;
      ldsA[(ml + 4) * 32 + cl] = (__bf16)f1.x;
      ldsA[(ml + 5) * 32 + cl] = (__bf16)f1.y;
      ldsA[(ml + 6) * 32 + cl] = (__bf16)f1.z;
      ldsA[(ml + 7) * 32 + cl] = (__bf16)f1.w;
      if (kk < 7) __builtin_prefetch(src + (size_t)32 * HW, 0, 0);  // global_prefetch_b8
    }
    __syncthreads();

    v16bf Bt[2][2];
#pragma unroll
    for (int nt = 0; nt < 2; ++nt) {
      const int n = n0 + nt * 16 + ln;
      Bt[0][nt] = packB_f32(wk + (size_t)n * CCH + kb + 16 * hi);
      Bt[1][nt] = packB_f32(wv + (size_t)n * CCH + kb + 16 * hi);
    }
#pragma unroll
    for (int mt = 0; mt < 4; ++mt) {
      v16bf A = packA_bf(ldsA + (mt * 16 + ln) * 32, 0, hi);
#pragma unroll
      for (int nt = 0; nt < 2; ++nt) {
        acc[0][nt][mt] = wmma_bf16(A, Bt[0][nt], acc[0][nt][mt]);
        acc[1][nt][mt] = wmma_bf16(A, Bt[1][nt], acc[1][nt][mt]);
      }
    }
  }

#pragma unroll
  for (int nt = 0; nt < 2; ++nt) {
    const int n = n0 + nt * 16 + ln;
    const float biask = bk[n], biasv = bv[n];
#pragma unroll
    for (int mt = 0; mt < 4; ++mt) {
      const int mb = m0 + mt * 16 + hi * 8;  // M = r + 8*(lane>=16)
#pragma unroll
      for (int r = 0; r < 8; ++r) {
        const size_t off = ((size_t)b * HW + (size_t)(mb + r)) * CCH + n;
        kout[off] = (__bf16)(acc[0][nt][mt][r] + biask);
        vout[off] = (__bf16)(acc[1][nt][mt][r] + biasv);
      }
    }
  }
}

// ---------------- Kernel C: per-ROI attention pooling (1 WG / ROI) ----------------

struct SampleCoord {
  int i00, i01, i10, i11;
  float w00, w01, w10, w11;
};

static __device__ __forceinline__ SampleCoord make_coord(float y1c, float x1c,
                                                         float dy, float dx, int p) {
  const int iy = p >> 4, ix = p & 15;
  float yy = y1c - 0.5f + ((float)iy + 0.5f) * dy;
  float xx = x1c - 0.5f + ((float)ix + 0.5f) * dx;
  float msk = (yy >= -1.0f && yy <= (float)FH && xx >= -1.0f && xx <= (float)FW) ? 1.0f : 0.0f;
  float yc = fminf(fmaxf(yy, 0.0f), (float)(FH - 1));
  float xc = fminf(fmaxf(xx, 0.0f), (float)(FW - 1));
  int y0 = (int)floorf(yc); int y1i = min(y0 + 1, FH - 1); float ly = yc - (float)y0;
  int x0 = (int)floorf(xc); int x1i = min(x0 + 1, FW - 1); float lx = xc - (float)x0;
  SampleCoord s;
  s.i00 = y0 * FW + x0;  s.i01 = y0 * FW + x1i;
  s.i10 = y1i * FW + x0; s.i11 = y1i * FW + x1i;
  s.w00 = msk * (1.0f - ly) * (1.0f - lx); s.w01 = msk * (1.0f - ly) * lx;
  s.w10 = msk * ly * (1.0f - lx);          s.w11 = msk * ly * lx;
  return s;
}

#define SMEM_RKV   0
#define SMEM_S     131072
#define SMEM_ATTNF (131072 + 65536)
#define SMEM_ATTNB (131072 + 65536 + 50176)
#define SMEM_TOTAL (131072 + 65536 + 50176 + 32768)  // 279552 B < 320 KB WGP LDS

__global__ __launch_bounds__(256) void roi_attn_kernel(
    const float* __restrict__ rois,
    const __bf16* __restrict__ kbuf, const __bf16* __restrict__ vbuf,  // [2,40000,256]
    const __bf16* __restrict__ qb,                                      // [64,256]
    const float* __restrict__ pos_attn,                                 // [49,256]
    const float* __restrict__ balance_p,
    float* __restrict__ out) {                                          // [1024,256,49]
  extern __shared__ char smem[];
  __bf16* rkv   = (__bf16*)(smem + SMEM_RKV);    // phase1: rk [l][c]; phase4: rv [c][l]
  float*  S     = (float*)(smem + SMEM_S);       // [64][256] scores
  float*  attnF = (float*)(smem + SMEM_ATTNF);   // [49][256] blended attention (f32)
  __bf16* attnB = (__bf16*)(smem + SMEM_ATTNB);  // [64][256] bf16, padded

  const int tid = threadIdx.x, lane = tid & 31, wid = tid >> 5;
  const int hi = lane >> 4, ln = lane & 15;
  const int roi_id = blockIdx.x;
  const float* roi = rois + (size_t)roi_id * 5;
  const int b = (int)roi[0];
  const float x1 = roi[1] * 0.25f, y1 = roi[2] * 0.25f;
  const float x2 = roi[3] * 0.25f, y2 = roi[4] * 0.25f;
  const float dy = (y2 - y1) * (1.0f / 16.0f), dx = (x2 - x1) * (1.0f / 16.0f);
  const float bal = balance_p[0];
  const size_t base = (size_t)b * HW;

  // init attention with position term
  for (int idx = tid; idx < NQ * 256; idx += 256) attnF[idx] = bal * pos_attn[idx];

  // phase 1: bilinear sample K -> rkv[l][c]; wave per point, lane covers 8 channels
  for (int p = wid; p < 256; p += 8) {
    SampleCoord s = make_coord(y1, x1, dy, dx, p);
    const __bf16* p00 = kbuf + (base + (size_t)s.i00) * CCH + lane * 8;
    const __bf16* p01 = kbuf + (base + (size_t)s.i01) * CCH + lane * 8;
    const __bf16* p10 = kbuf + (base + (size_t)s.i10) * CCH + lane * 8;
    const __bf16* p11 = kbuf + (base + (size_t)s.i11) * CCH + lane * 8;
    v8bf a0 = *(const v8bf*)p00, a1 = *(const v8bf*)p01;
    v8bf a2 = *(const v8bf*)p10, a3 = *(const v8bf*)p11;
    v8bf o;
#pragma unroll
    for (int i = 0; i < 8; ++i)
      o[i] = (__bf16)(s.w00 * (float)a0[i] + s.w01 * (float)a1[i] +
                      s.w10 * (float)a2[i] + s.w11 * (float)a3[i]);
    *(v8bf*)(rkv + (size_t)p * CCH + lane * 8) = o;
  }
  __syncthreads();

  const float wBal = (1.0f - bal) * 0.25f;  // mean over 4 heads folded in

  // phase 2: per-head scores (WMMA) + softmax + accumulate
  for (int h = 0; h < NHEADS; ++h) {
#pragma unroll
    for (int i = 0; i < 8; ++i) {
      const int t = wid * 8 + i;          // 64 tiles of 16x16 over [q=64][l=256]
      const int mt = t >> 4, nt = t & 15;
      const __bf16* qrow = qb  + (size_t)(mt * 16 + ln) * CCH + h * HDIM;
      const __bf16* krow = rkv + (size_t)(nt * 16 + ln) * CCH + h * HDIM;
      v8f acc = {0, 0, 0, 0, 0, 0, 0, 0};
      acc = wmma_bf16(packA_bf(qrow, 0, hi),  packB_bf(krow, 0, hi),  acc);
      acc = wmma_bf16(packA_bf(qrow, 32, hi), packB_bf(krow, 32, hi), acc);
      const int q0 = mt * 16 + hi * 8;
      const int l  = nt * 16 + ln;
#pragma unroll
      for (int r = 0; r < 8; ++r) S[(q0 + r) * 256 + l] = acc[r] * 0.0625f;  // *C^-0.5
    }
    __syncthreads();
    for (int row = wid; row < NQ; row += 8) {
      const float* sr = S + row * 256 + lane * 8;
      float v[8]; float mx = -3.0e38f;
#pragma unroll
      for (int i = 0; i < 8; ++i) { v[i] = sr[i]; mx = fmaxf(mx, v[i]); }
      mx = wredMax(mx);
      float sm = 0.0f;
#pragma unroll
      for (int i = 0; i < 8; ++i) { v[i] = expf(v[i] - mx); sm += v[i]; }
      sm = wredSum(sm);
      const float sc = wBal / sm;
      float* ar = attnF + row * 256 + lane * 8;
#pragma unroll
      for (int i = 0; i < 8; ++i) ar[i] += v[i] * sc;
    }
    __syncthreads();
  }

  // phase 3: attention -> bf16 B-matrix (rows 49..63 zero)
  for (int idx = tid; idx < 64 * 256; idx += 256) {
    const int row = idx >> 8;
    attnB[idx] = (__bf16)((row < NQ) ? attnF[idx] : 0.0f);
  }
  __syncthreads();

  // phase 4: bilinear sample V -> rkv[c][l] (reuse LDS, rk no longer needed)
  for (int p = wid; p < 256; p += 8) {
    SampleCoord s = make_coord(y1, x1, dy, dx, p);
    const __bf16* p00 = vbuf + (base + (size_t)s.i00) * CCH + lane * 8;
    const __bf16* p01 = vbuf + (base + (size_t)s.i01) * CCH + lane * 8;
    const __bf16* p10 = vbuf + (base + (size_t)s.i10) * CCH + lane * 8;
    const __bf16* p11 = vbuf + (base + (size_t)s.i11) * CCH + lane * 8;
    v8bf a0 = *(const v8bf*)p00, a1 = *(const v8bf*)p01;
    v8bf a2 = *(const v8bf*)p10, a3 = *(const v8bf*)p11;
#pragma unroll
    for (int i = 0; i < 8; ++i) {
      float fv = s.w00 * (float)a0[i] + s.w01 * (float)a1[i] +
                 s.w10 * (float)a2[i] + s.w11 * (float)a3[i];
      rkv[(size_t)(lane * 8 + i) * 256 + p] = (__bf16)fv;
    }
  }
  __syncthreads();

  // phase 5: out[c][q] = sum_l rv[c][l] * attn[q][l]  (M=256, N=64, K=256)
#pragma unroll
  for (int i = 0; i < 8; ++i) {
    const int t = wid * 8 + i;
    const int ct = t >> 2, nt = t & 3;
    const __bf16* arow = rkv   + (size_t)(ct * 16 + ln) * 256;
    const __bf16* brow = attnB + (size_t)(nt * 16 + ln) * 256;
    v8f acc = {0, 0, 0, 0, 0, 0, 0, 0};
#pragma unroll
    for (int kb = 0; kb < 256; kb += 32)
      acc = wmma_bf16(packA_bf(arow, kb, hi), packB_bf(brow, kb, hi), acc);
    const int q = nt * 16 + ln;
    if (q < NQ) {
      const int c0 = ct * 16 + hi * 8;
      float* op = out + ((size_t)roi_id * CCH + c0) * NQ + q;
#pragma unroll
      for (int r = 0; r < 8; ++r) op[(size_t)r * NQ] = acc[r];
    }
  }
}

// ---------------- Host launcher ----------------

extern "C" void kernel_launch(void* const* d_in, const int* in_sizes, int n_in,
                              void* d_out, int out_size, void* d_ws, size_t ws_size,
                              hipStream_t stream) {
  (void)in_sizes; (void)n_in; (void)out_size; (void)ws_size;
  const float* features = (const float*)d_in[0];
  const float* rois     = (const float*)d_in[1];
  const float* w_k      = (const float*)d_in[2];
  const float* b_k      = (const float*)d_in[3];
  const float* w_v      = (const float*)d_in[4];
  const float* b_v      = (const float*)d_in[5];
  const float* queries  = (const float*)d_in[6];
  const float* pos_w    = (const float*)d_in[7];
  const float* braw     = (const float*)d_in[8];

  char* ws = (char*)d_ws;
  float*  balance  = (float*)ws;                         // 4 B
  float*  pos_attn = (float*)(ws + 256);                 // 50176 B
  __bf16* qb       = (__bf16*)(ws + 51200);              // 32768 B
  __bf16* kbuf     = (__bf16*)(ws + 131072);             // 40,960,000 B
  __bf16* vbuf     = (__bf16*)(ws + 131072 + 40960000);  // 40,960,000 B
  float* out = (float*)d_out;

  (void)hipFuncSetAttribute((const void*)roi_attn_kernel,
                            hipFuncAttributeMaxDynamicSharedMemorySize, SMEM_TOTAL);

  prep_kernel<<<1, 256, 0, stream>>>(pos_w, queries, braw, pos_attn, qb, balance);
  conv_kv_kernel<<<2 * 625, 256, 0, stream>>>(features, w_k, b_k, w_v, b_v, kbuf, vbuf);
  roi_attn_kernel<<<1024, 256, SMEM_TOTAL, stream>>>(rois, kbuf, vbuf, qb, pos_attn,
                                                     balance, out);
}
